// LinearModel_21320217657809
// MI455X (gfx1250) — matrix operand order repaired and checked by reading, not compiled, not run on hardware
//
#include <hip/hip_runtime.h>

// MI455X / gfx1250 implementation.
// Phase 1: split-precision bf16 WMMA GEMM computing the sliding correlation
//          S[b,i,d] = sum_t y[b,t,d] * w[t-i,d]  (t>=i)   -- 17.2 GFLOP
// Phase 2: per-chain autoregressive recurrence
//          out_i = bias + S_i + sum_{j<i} out_j * w[2048-i+j]
// Workspace (assumed >= 160 MiB):
//   [0,64MiB)      A_hi  bf16  [d][b][t]
//   [64,128MiB)    A_lo  bf16  [d][b][t]
//   [128,144MiB)   S     f32   [d][b][i]
//   [144,160MiB)   hist  f32   [i][chain]  (chain = b*32+d)

typedef __bf16 bf16;
typedef __bf16 v16bf __attribute__((ext_vector_type(16)));
typedef float  v8f   __attribute__((ext_vector_type(8)));

#define NB   512
#define LEN  2048
#define DY   32
#define PRED 256

// padded weight column: 256 leading zeros + 2048 entries = 2304 bf16 = 1152 dw
#define WCOL  2304
#define WDW   (WCOL / 2 + 4)

// ---------------------------------------------------------------------------
// Kernel 1: pack y_c [b][t][d] fp32 -> A_hi/A_lo bf16 [d][b][t] (hi/lo split)
// ---------------------------------------------------------------------------
__global__ __launch_bounds__(256) void pack_kernel(const float* __restrict__ y,
                                                   bf16* __restrict__ Ah,
                                                   bf16* __restrict__ Al) {
    __shared__ float tile[256][33];   // pad 33: stride gcd(33,64)=1 -> conflict free
    const int b   = blockIdx.x >> 3;
    const int tc  = (blockIdx.x & 7) * 256;
    const int tid = threadIdx.x;
    const int d0  = tid & 31, r0 = tid >> 5;

    const float* src = y + (size_t)b * (LEN * DY) + (size_t)tc * DY;
#pragma unroll
    for (int it = 0; it < 32; ++it) {
        int t = r0 + it * 8;
        tile[t][d0] = src[t * DY + d0];       // coalesced over d
    }
    __syncthreads();
#pragma unroll
    for (int d = 0; d < DY; ++d) {
        float x  = tile[tid][d];
        bf16 hi  = (bf16)x;
        bf16 lo  = (bf16)(x - (float)hi);
        size_t o = ((size_t)d * NB + b) * LEN + tc + tid;   // coalesced over t
        Ah[o] = hi;
        Al[o] = lo;
    }
}

// ---------------------------------------------------------------------------
// Kernel 2: per-d GEMM  S[d] (512x256) = A[d] (512x2048) * Toeplitz(w[:,d])
// WG = 256 thr (8 waves), WG tile 128(M)x128(N), wave tile 64x32, K-chunk 32.
// bf16 split: acc += Ahi*Bhi + Alo*Bhi + Ahi*Blo  (f32 accumulate)
// Weight column pre-split to bf16 hi/lo in LDS once (two parity-shifted
// copies -> B fragments are pure aligned ds_load_b32, no VALU in hot loop).
// ---------------------------------------------------------------------------
__global__ __launch_bounds__(256) void gemm_kernel(const bf16* __restrict__ Ah,
                                                   const bf16* __restrict__ Al,
                                                   const float* __restrict__ w,
                                                   float* __restrict__ S) {
    __shared__ bf16 aHi[128][40];    // pad to 40 -> conflict-free b128 frag loads
    __shared__ bf16 aLo[128][40];
    __shared__ unsigned int wHiU[2][WDW];  // [copy][dword]; copy1 shifted by 1 elem
    __shared__ unsigned int wLoU[2][WDW];

    const int bx  = blockIdx.x;
    const int d   = bx >> 3;
    const int mb  = (bx >> 1) & 3;    // 4 M-tiles of 128
    const int nb  = bx & 1;           // 2 N-tiles of 128
    const int b0  = mb * 128;
    const int i0  = nb * 128;
    const int tid = threadIdx.x;
    const int wid = tid >> 5, lane = tid & 31;
    const int wm  = wid & 1;          // 2 waves over M (64 each)
    const int wn  = wid >> 1;         // 4 waves over N (32 each)
    const int nlane = lane & 15;
    const int hlf   = lane >> 4;

    // ---- stage padded, parity-duplicated bf16 hi/lo weight column (once) ----
    {
        bf16* h0 = (bf16*)&wHiU[0][0];
        bf16* h1 = (bf16*)&wHiU[1][0];
        bf16* l0 = (bf16*)&wLoU[0][0];
        bf16* l1 = (bf16*)&wLoU[1][0];
        for (int idx = tid; idx < WCOL; idx += 256) {
            int t  = idx - 256;                       // column index (pad < 0)
            float x = (t >= 0) ? w[t * DY + d] : 0.0f;
            bf16 hb = (bf16)x;
            bf16 lb = (bf16)(x - (float)hb);
            h0[idx] = hb; l0[idx] = lb;
            if (idx > 0) { h1[idx - 1] = hb; l1[idx - 1] = lb; }
        }
        if (tid == 0) {
            h1[WCOL - 1] = (bf16)0.0f;
            l1[WCOL - 1] = (bf16)0.0f;
        }
    }

    v8f acc[4][2];
#pragma unroll
    for (int fm = 0; fm < 4; ++fm)
#pragma unroll
        for (int fn = 0; fn < 2; ++fn) acc[fm][fn] = (v8f)0.0f;

    const bf16* ahg = Ah + ((size_t)d * NB + b0) * LEN;
    const bf16* alg = Al + ((size_t)d * NB + b0) * LEN;

    const int am = tid >> 1;            // staging row 0..127
    const int ak = (tid & 1) * 16;      // staging k-half

    for (int k0 = 0; k0 < LEN; k0 += 32) {
        __syncthreads();
        // ---- stage A hi/lo tile (128 x 32) via 2x16B per thread ----
        {
            const uint4* ph = (const uint4*)(ahg + (size_t)am * LEN + k0 + ak);
            const uint4* pl = (const uint4*)(alg + (size_t)am * LEN + k0 + ak);
            uint4 h0 = ph[0], h1 = ph[1];
            uint4 l0 = pl[0], l1 = pl[1];
            *(uint4*)(&aHi[am][ak])     = h0;
            *(uint4*)(&aHi[am][ak + 8]) = h1;
            *(uint4*)(&aLo[am][ak])     = l0;
            *(uint4*)(&aLo[am][ak + 8]) = l1;
            if (k0 + 32 < LEN)
                __builtin_prefetch(ahg + (size_t)am * LEN + k0 + 32 + ak, 0, 0);
        }
        __syncthreads();

        // ---- B fragments: aligned b32 loads from pre-split weight column ----
        // B layout (32x16 K-major): lane n = lane&15, element e -> k = hlf*16+e.
        // col element for (k,n): 256 + (k0 + hlf*16 + e) - (i0 + n)
        v16bf bHi[2], bLo[2];
#pragma unroll
        for (int fn = 0; fn < 2; ++fn) {
            int n    = wn * 32 + fn * 16 + nlane;       // 0..127
            int base = 256 + k0 + hlf * 16 - i0 - n;    // padded elem idx of e=0
            int c    = base & 1;                        // parity-matched copy
            int dw   = base >> 1;                       // aligned dword start
            union Uv { unsigned int u[8]; v16bf v; } th, tl;
#pragma unroll
            for (int q = 0; q < 8; ++q) {
                th.u[q] = wHiU[c][dw + q];
                tl.u[q] = wLoU[c][dw + q];
            }
            bHi[fn] = th.v;
            bLo[fn] = tl.v;
        }

        // ---- load A fragments (16x32, same contiguous k order as B) ----
        // lane m = lane&15 ; element e -> k = hlf*16+e
        v16bf aH[4], aL[4];
        const int arow  = wm * 64 + nlane;
        const int akoff = hlf * 16;
#pragma unroll
        for (int fm = 0; fm < 4; ++fm) {
            int m = arow + fm * 16;
            union U { uint4 u[2]; v16bf v; } th, tl;
            th.u[0] = *(const uint4*)(&aHi[m][akoff]);
            th.u[1] = *(const uint4*)(&aHi[m][akoff + 8]);
            tl.u[0] = *(const uint4*)(&aLo[m][akoff]);
            tl.u[1] = *(const uint4*)(&aLo[m][akoff + 8]);
            aH[fm] = th.v;
            aL[fm] = tl.v;
        }

        // ---- 24 WMMAs: split-precision accumulation ----
#pragma unroll
        for (int fm = 0; fm < 4; ++fm)
#pragma unroll
            for (int fn = 0; fn < 2; ++fn) {
                v8f c = acc[fm][fn];
                c = __builtin_amdgcn_wmma_f32_16x16x32_bf16(
                        false, aH[fm], false, bHi[fn], (short)0, c, false, false);
                c = __builtin_amdgcn_wmma_f32_16x16x32_bf16(
                        false, aL[fm], false, bHi[fn], (short)0, c, false, false);
                c = __builtin_amdgcn_wmma_f32_16x16x32_bf16(
                        false, aH[fm], false, bLo[fn], (short)0, c, false, false);
                acc[fm][fn] = c;
            }
    }

    // ---- store C: lane (r) -> m = r + hlf*8, n = lane&15 ----
#pragma unroll
    for (int fm = 0; fm < 4; ++fm)
#pragma unroll
        for (int fn = 0; fn < 2; ++fn) {
            int mrow = b0 + wm * 64 + fm * 16 + hlf * 8;
            int ncol = i0 + wn * 32 + fn * 16 + nlane;
            float* dst = S + ((size_t)d * NB + mrow) * PRED + ncol;
#pragma unroll
            for (int r = 0; r < 8; ++r) dst[(size_t)r * PRED] = acc[fm][fn][r];
        }
}

// ---------------------------------------------------------------------------
// Kernel 3: recurrence. 128 threads/block = 128 chains (4 b x 32 d).
// i-tiles of 8: cross part uses rolling 15-wide weight window (LDS) and
// coalesced global history [i][chain]; in-tile 8x8 triangular part in regs.
// ---------------------------------------------------------------------------
__global__ __launch_bounds__(128) void recur_kernel(const float* __restrict__ S,
                                                    const float* __restrict__ w,
                                                    const float* __restrict__ bias,
                                                    float* __restrict__ out,
                                                    float* __restrict__ hist) {
    __shared__ float wl[32 * 261 + 16];   // wl[d*261 + r] = w[1792+r][d], padded

    const int tid   = threadIdx.x;
    const int d     = tid & 31;
    const int bb    = tid >> 5;
    const int b     = blockIdx.x * 4 + bb;
    const int chain = blockIdx.x * 128 + tid;   // = b*32 + d

    for (int idx = tid; idx < 32 * 256; idx += 128) {
        int dd = idx >> 8, r = idx & 255;
        wl[dd * 261 + r] = w[(1792 + r) * DY + dd];
    }
    __syncthreads();

    const float bs = bias[d];
    float wt[8];                       // in-tile weights w[2048 - delta][d]
#pragma unroll
    for (int del = 1; del <= 7; ++del) wt[del] = w[(2048 - del) * DY + d];
    wt[0] = 0.0f;

    const float* Sp  = S + ((size_t)d * NB + b) * PRED;
    float*       op  = out + (size_t)b * PRED * DY + d;
    const float* wld = wl + d * 261;

    for (int i0 = 0; i0 < PRED; i0 += 8) {
        float acc[8] = {0.f, 0.f, 0.f, 0.f, 0.f, 0.f, 0.f, 0.f};

        if (i0 > 0) {
            const int rb = 256 - i0;
            float wv[15];                       // wv[s] = W(rb-7+s+j)
#pragma unroll
            for (int s = 0; s < 15; ++s) wv[s] = wld[rb - 7 + s];

            for (int j = 0; j < i0; j += 8) {
                float h[8];
#pragma unroll
                for (int t = 0; t < 8; ++t)
                    h[t] = hist[(size_t)(j + t) * (NB * DY) + chain]; // coalesced
#pragma unroll
                for (int t = 0; t < 8; ++t)
#pragma unroll
                    for (int p = 0; p < 8; ++p)
                        acc[p] = fmaf(h[t], wv[7 - p + t], acc[p]);
                // slide window by 8 (pad in wl covers the final dead reload)
#pragma unroll
                for (int s = 0; s < 7; ++s) wv[s] = wv[s + 8];
#pragma unroll
                for (int s = 7; s < 15; ++s) wv[s] = wld[rb - 7 + s + j + 8];
            }
        }

        float4 s0 = *(const float4*)(Sp + i0);
        float4 s1 = *(const float4*)(Sp + i0 + 4);
        float Sv[8] = {s0.x, s0.y, s0.z, s0.w, s1.x, s1.y, s1.z, s1.w};

        float o[8];
#pragma unroll
        for (int p = 0; p < 8; ++p) {
            float v = bs + Sv[p] + acc[p];
#pragma unroll
            for (int q = 0; q < 8; ++q)
                if (q < p) v = fmaf(o[q], wt[p - q], v);
            o[p] = v;
            hist[(size_t)(i0 + p) * (NB * DY) + chain] = v;  // coalesced
            op[(size_t)(i0 + p) * DY] = v;                   // coalesced over d
        }
    }
}

// ---------------------------------------------------------------------------
extern "C" void kernel_launch(void* const* d_in, const int* in_sizes, int n_in,
                              void* d_out, int out_size, void* d_ws, size_t ws_size,
                              hipStream_t stream) {
    const float* y_c  = (const float*)d_in[0];
    const float* wts  = (const float*)d_in[1];
    const float* bias = (const float*)d_in[2];
    // d_in[3] = pred_len (scalar) -- baked into PRED.

    float* out = (float*)d_out;
    char*  ws  = (char*)d_ws;
    bf16*  Ah   = (bf16*)(ws);
    bf16*  Al   = (bf16*)(ws + (size_t)64 * 1024 * 1024);
    float* S    = (float*)(ws + (size_t)128 * 1024 * 1024);
    float* hist = (float*)(ws + (size_t)144 * 1024 * 1024);

    pack_kernel<<<4096, 256, 0, stream>>>(y_c, Ah, Al);
    gemm_kernel<<<256, 256, 0, stream>>>(Ah, Al, wts, S);
    recur_kernel<<<128, 128, 0, stream>>>(S, wts, bias, out, hist);
}
